// Attention_63831803953511
// MI455X (gfx1250) — matrix operand order repaired and checked by reading, not compiled, not run on hardware
//
#include <hip/hip_runtime.h>
#include <hip/hip_bf16.h>

// ---------------------------------------------------------------------------
// Transformer block on gfx1250: LN1 -> QKV -> flash attention -> out-proj+res
// -> LN2 -> FFN(relu)+res.  All GEMM-shaped math on v_wmma_f32_16x16x32_bf16.
// GEMM uses double-buffered GLOBAL_LOAD_ASYNC_TO_LDS_B128 staging (ASYNCcnt).
// ---------------------------------------------------------------------------

#define B_ 2
#define T_ 2048
#define E_ 1024
#define H_ 16
#define D_ 64
#define F_ 4096          // 4*E
#define M_ (B_ * T_)     // 4096 rows of activations
#define BK_ 64           // GEMM K panel depth

typedef __bf16 bf16_t;
typedef __attribute__((ext_vector_type(16))) __bf16 v16bf;
typedef __attribute__((ext_vector_type(8)))  __bf16 v8bf;
typedef __attribute__((ext_vector_type(8)))  float  v8f;

// ---------------------------------------------------------------------------
// WMMA fragment helpers (layouts per CDNA5 ISA §7.12.2, wave32)
// ---------------------------------------------------------------------------

// A-matrix 16x32 bf16 fragment from a row-major [16 x >=32] source (lda elems).
//   lane l<16 : row=l, elems[0..7]=K0..7,  elems[8..15]=K16..23
//   lane l>=16: row=l-16, elems[0..7]=K8..15, elems[8..15]=K24..31
__device__ inline v16bf load_a_frag(const bf16_t* base, int lda) {
  int l = threadIdx.x & 31;
  int r = l & 15, half = l >> 4;
  const bf16_t* p = base + (size_t)r * lda + half * 8;
  v8bf lo = *(const v8bf*)(p);
  v8bf hi = *(const v8bf*)(p + 16);
  v16bf out;
#pragma unroll
  for (int i = 0; i < 8; ++i) { out[i] = lo[i]; out[i + 8] = hi[i]; }
  return out;
}

// B-matrix 32x16 bf16 fragment from a TRANSPOSED row-major [16 x >=32] source
// (Bt[n][k], ldb elems):  lane l<16: col=l, K0..7,K16..23; lane l>=16: col=l-16, K8..15,K24..31.
__device__ inline v16bf load_bt_frag(const bf16_t* base, int ldb) {
  int l = threadIdx.x & 31;
  int col = l & 15, half = l >> 4;
  const bf16_t* p = base + (size_t)col * ldb + half * 8;
  v8bf lo = *(const v8bf*)(p);
  v8bf hi = *(const v8bf*)(p + 16);
  v16bf out;
#pragma unroll
  for (int i = 0; i < 8; ++i) { out[i] = lo[i]; out[i + 8] = hi[i]; }
  return out;
}

__device__ inline v8f wmma_bf16(v16bf a, v16bf b, v8f c) {
  return __builtin_amdgcn_wmma_f32_16x16x32_bf16(false, a, false, b,
                                                 (short)0, c, false, false);
}

// ---------------------------------------------------------------------------
// Async global -> LDS copy helpers (CDNA5 ISA).
// INST_OFFSET is added to BOTH the LDS and the global address, so one base
// address pair serves several 16B issues.
// ---------------------------------------------------------------------------

typedef __attribute__((address_space(3))) bf16_t lds_bf16_t;

__device__ inline unsigned lds_addr_of(bf16_t* p) {
  return (unsigned)(unsigned long long)(lds_bf16_t*)p;  // AS3 offset
}

// 64 bytes per lane: four async b128 issues at offsets 0/16/32/48.
__device__ inline void async_copy_64B(unsigned lds, const bf16_t* g) {
  unsigned long long ga = (unsigned long long)(size_t)g;
  asm volatile(
      "global_load_async_to_lds_b128 %0, %1, off\n\t"
      "global_load_async_to_lds_b128 %0, %1, off offset:16\n\t"
      "global_load_async_to_lds_b128 %0, %1, off offset:32\n\t"
      "global_load_async_to_lds_b128 %0, %1, off offset:48"
      :: "v"(lds), "v"(ga) : "memory");
}

// 32 bytes per lane: two async b128 issues at offsets 0/16.
__device__ inline void async_copy_32B(unsigned lds, const bf16_t* g) {
  unsigned long long ga = (unsigned long long)(size_t)g;
  asm volatile(
      "global_load_async_to_lds_b128 %0, %1, off\n\t"
      "global_load_async_to_lds_b128 %0, %1, off offset:16"
      :: "v"(lds), "v"(ga) : "memory");
}

__device__ inline void wait_async_le6() {
  asm volatile("s_wait_asynccnt 0x6" ::: "memory");
}
__device__ inline void wait_async_0() {
  asm volatile("s_wait_asynccnt 0x0" ::: "memory");
}

// ---------------------------------------------------------------------------
// Weight conversion kernels (fp32 -> bf16, transposed to [N][K])
// ---------------------------------------------------------------------------

__global__ void transpose_to_bf16(const float* __restrict__ src,
                                  bf16_t* __restrict__ dst, int K, int N) {
  int n = blockIdx.x;
  for (int k = threadIdx.x; k < K; k += blockDim.x)
    dst[(size_t)n * K + k] = (bf16_t)src[(size_t)k * N + n];
}

// Wq/Wk/Wv are [H,E,D]; build Wqkv_t[3072][1024] with row n = output channel.
__global__ void qkv_weights_to_bf16(const float* __restrict__ Wq,
                                    const float* __restrict__ Wk,
                                    const float* __restrict__ Wv,
                                    bf16_t* __restrict__ dst) {
  int n = blockIdx.x;                 // 0..3071
  int which = n >> 10, nn = n & 1023;
  int h = nn >> 6, d = nn & 63;
  const float* W = (which == 0) ? Wq : (which == 1) ? Wk : Wv;
  for (int e = threadIdx.x; e < E_; e += blockDim.x)
    dst[(size_t)n * E_ + e] = (bf16_t)W[((size_t)h * E_ + e) * D_ + d];
}

// ---------------------------------------------------------------------------
// LayerNorm: fp32 [row][E] -> bf16 [row][E]
// ---------------------------------------------------------------------------

__global__ __launch_bounds__(256) void layernorm_bf16(
    const float* __restrict__ x, const float* __restrict__ g,
    const float* __restrict__ b, bf16_t* __restrict__ out) {
  int row = blockIdx.x;
  const float* xr = x + (size_t)row * E_;
  float s = 0.f, s2 = 0.f;
  for (int i = threadIdx.x; i < E_; i += 256) {
    float v = xr[i]; s += v; s2 += v * v;
  }
#pragma unroll
  for (int m = 16; m > 0; m >>= 1) {
    s  += __shfl_xor(s,  m, 32);
    s2 += __shfl_xor(s2, m, 32);
  }
  __shared__ float rs[8], rs2[8];
  __shared__ float mean_s, rstd_s;
  int wave = threadIdx.x >> 5;
  if ((threadIdx.x & 31) == 0) { rs[wave] = s; rs2[wave] = s2; }
  __syncthreads();
  if (threadIdx.x == 0) {
    float a = 0.f, a2 = 0.f;
    for (int w = 0; w < 8; ++w) { a += rs[w]; a2 += rs2[w]; }
    float mean = a / (float)E_;
    float var  = a2 / (float)E_ - mean * mean;
    mean_s = mean;
    rstd_s = rsqrtf(var + 1e-5f);
  }
  __syncthreads();
  float mean = mean_s, rstd = rstd_s;
  bf16_t* orow = out + (size_t)row * E_;
  for (int i = threadIdx.x; i < E_; i += 256)
    orow[i] = (bf16_t)((xr[i] - mean) * rstd * g[i] + b[i]);
}

// ---------------------------------------------------------------------------
// Generic WMMA GEMM: C[M,N] = A[M,K] (bf16) * Bt[N,K]^T (bf16)
// Block tile 128x64, 8 waves (4x2), each wave 32x32 (2x2 WMMA tiles).
// K panel 64, double-buffered in LDS via async global->LDS copies:
//   issue panel kp+1, s_wait_asynccnt 6 (panel kp landed), barrier, 8 WMMAs,
//   barrier (panel kp+2's buffer free to overwrite).
// mode 1: split epilogue -> q[B,H,T,D], k[B,H,T,D], vT[B,H,D,T] (bf16)
// mode 2: outf = acc + bias[col] + resid (fp32)
// mode 3: outb = bf16(relu(acc + bias[col]))
// ---------------------------------------------------------------------------

__global__ __launch_bounds__(256) void gemm_wmma(
    const bf16_t* __restrict__ A, const bf16_t* __restrict__ Bt,
    int N, int K, const float* __restrict__ bias,
    const float* __restrict__ resid, float* __restrict__ outf,
    bf16_t* __restrict__ outb, bf16_t* __restrict__ qo,
    bf16_t* __restrict__ ko, bf16_t* __restrict__ vo, int mode) {
  __shared__ __align__(16) bf16_t As[2][128 * BK_];  // 2 x 16KB
  __shared__ __align__(16) bf16_t Bs[2][64 * BK_];   // 2 x 8KB
  int tid = threadIdx.x;
  int lane = tid & 31, wave = tid >> 5;
  int wm = wave & 3, wn = wave >> 2;
  int rowBase = blockIdx.y * 128;
  int colBase = blockIdx.x * 64;

  // Per-thread staging slices:
  //   A panel 128x64: thread -> row tid>>1, 32 elems (64B) at (tid&1)*32.
  //   B panel  64x64: thread -> row tid>>2, 16 elems (32B) at (tid&3)*16.
  int ar = tid >> 1, ac = (tid & 1) * 32;
  int bn = tid >> 2, bc = (tid & 3) * 16;
  const bf16_t* gA = A + (size_t)(rowBase + ar) * K + ac;
  const bf16_t* gB = Bt + (size_t)(colBase + bn) * K + bc;
  unsigned ldsA[2], ldsB[2];
#pragma unroll
  for (int b = 0; b < 2; ++b) {
    ldsA[b] = lds_addr_of(&As[b][ar * BK_ + ac]);
    ldsB[b] = lds_addr_of(&Bs[b][bn * BK_ + bc]);
  }

  v8f acc[2][2] = {};

  int nk = K / BK_;
  // Prologue: stage panel 0 into buffer 0 (6 async issues per thread).
  async_copy_64B(ldsA[0], gA);
  async_copy_32B(ldsB[0], gB);

  for (int kp = 0; kp < nk; ++kp) {
    int buf = kp & 1;
    if (kp + 1 < nk) {
      async_copy_64B(ldsA[buf ^ 1], gA + (size_t)(kp + 1) * BK_);
      async_copy_32B(ldsB[buf ^ 1], gB + (size_t)(kp + 1) * BK_);
      wait_async_le6();  // only next panel's 6 issues may remain in flight
    } else {
      wait_async_0();
    }
    __syncthreads();  // all waves' panel-kp data visible in LDS

    const bf16_t* Asb = As[buf];
    const bf16_t* Bsb = Bs[buf];
#pragma unroll
    for (int kc = 0; kc < BK_; kc += 32) {
      v16bf bfrag[2];
#pragma unroll
      for (int j = 0; j < 2; ++j)
        bfrag[j] = load_bt_frag(Bsb + (wn * 32 + j * 16) * BK_ + kc, BK_);
#pragma unroll
      for (int i = 0; i < 2; ++i) {
        v16bf a = load_a_frag(Asb + (wm * 32 + i * 16) * BK_ + kc, BK_);
#pragma unroll
        for (int j = 0; j < 2; ++j)
          acc[i][j] = wmma_bf16(a, bfrag[j], acc[i][j]);
      }
    }
    __syncthreads();  // buffer free before panel kp+2 overwrites it
  }

  // Epilogue: C layout (§7.12.2): VGPR v -> row half*8+v (of tile), lane%16 -> col.
  int col16 = lane & 15, half = lane >> 4;
#pragma unroll
  for (int i = 0; i < 2; ++i) {
#pragma unroll
    for (int j = 0; j < 2; ++j) {
      int row0 = rowBase + wm * 32 + i * 16 + half * 8;
      int col  = colBase + wn * 32 + j * 16 + col16;
#pragma unroll
      for (int v = 0; v < 8; ++v) {
        int row = row0 + v;
        float val = acc[i][j][v];
        if (mode == 1) {
          int which = col >> 10, nn = col & 1023;
          int hh = nn >> 6, dd = nn & 63;
          int bb = row >> 11, tt = row & (T_ - 1);
          size_t qk = (((size_t)bb * H_ + hh) * T_ + tt) * D_ + dd;
          if (which == 0)      qo[qk] = (bf16_t)val;
          else if (which == 1) ko[qk] = (bf16_t)val;
          else vo[(((size_t)bb * H_ + hh) * D_ + dd) * T_ + tt] = (bf16_t)val;
        } else if (mode == 2) {
          size_t idx = (size_t)row * N + col;
          outf[idx] = val + bias[col] + resid[idx];
        } else {  // mode 3: bias + relu, bf16 out
          float t = val + bias[col];
          outb[(size_t)row * N + col] = (bf16_t)(t > 0.f ? t : 0.f);
        }
      }
    }
  }
}

// ---------------------------------------------------------------------------
// Flash attention: one wave per 16-row Q tile; 32-key chunks; online softmax.
// Q,K: [B,H,T,D] bf16.  Vt: [B,H,D,T] bf16.  Out -> concat[B*T][E] bf16.
// scale = E^-0.5 (matches reference, which scales by embed size).
// ---------------------------------------------------------------------------

__global__ __launch_bounds__(32) void flash_attn(
    const bf16_t* __restrict__ Q, const bf16_t* __restrict__ Kc,
    const bf16_t* __restrict__ Vt, bf16_t* __restrict__ concat) {
  int tile = blockIdx.x;               // B*H*(T/16) tiles
  int qt = tile & (T_ / 16 - 1);
  int bh = tile >> 7;
  int bb = bh >> 4, hh = bh & 15;
  int q0 = qt << 4;

  const bf16_t* Qb = Q  + ((size_t)bh * T_ + q0) * D_;
  const bf16_t* Kb = Kc + (size_t)bh * T_ * D_;
  const bf16_t* Vb = Vt + (size_t)bh * D_ * T_;

  __shared__ __align__(16) bf16_t Pld[16 * 32];

  int lane = threadIdx.x & 31;
  int col = lane & 15, half = lane >> 4;
  const float scale = 0.03125f;  // 1024^-0.5

  v8f o0 = {}, o1 = {}, o2 = {}, o3 = {};
  float mrun[8], lrun[8];
#pragma unroll
  for (int i = 0; i < 8; ++i) { mrun[i] = -3.0e38f; lrun[i] = 0.f; }

  // Q fragments are loop-invariant: 16x64 = two 16x32 A fragments.
  v16bf qa0 = load_a_frag(Qb, D_);
  v16bf qa1 = load_a_frag(Qb + 32, D_);

  for (int s0 = 0; s0 <= q0 + 15; s0 += 32) {
    // S = Q * K^T for 32 keys -> two 16x16 fp32 accumulators.
    v8f sc0 = {}, sc1 = {};
    {
      v16bf b00 = load_bt_frag(Kb + (size_t)s0 * D_, D_);
      v16bf b01 = load_bt_frag(Kb + (size_t)s0 * D_ + 32, D_);
      sc0 = wmma_bf16(qa0, b00, sc0);
      sc0 = wmma_bf16(qa1, b01, sc0);
      v16bf b10 = load_bt_frag(Kb + (size_t)(s0 + 16) * D_, D_);
      v16bf b11 = load_bt_frag(Kb + (size_t)(s0 + 16) * D_ + 32, D_);
      sc1 = wmma_bf16(qa0, b10, sc1);
      sc1 = wmma_bf16(qa1, b11, sc1);
    }

    int key0 = s0 + col, key1 = s0 + 16 + col;
#pragma unroll
    for (int i = 0; i < 8; ++i) {
      int rowg = q0 + half * 8 + i;
      float v0 = sc0[i] * scale; if (key0 > rowg) v0 = -3.0e38f;
      float v1 = sc1[i] * scale; if (key1 > rowg) v1 = -3.0e38f;

      // row-wise max over the 16 N-lanes of this half (xor masks stay in-half)
      float cm = fmaxf(v0, v1);
#pragma unroll
      for (int m = 1; m < 16; m <<= 1) cm = fmaxf(cm, __shfl_xor(cm, m, 32));
      float nm   = fmaxf(mrun[i], cm);
      float corr = __expf(mrun[i] - nm);
      float p0 = __expf(v0 - nm), p1 = __expf(v1 - nm);
      float rsum = p0 + p1;
#pragma unroll
      for (int m = 1; m < 16; m <<= 1) rsum += __shfl_xor(rsum, m, 32);
      lrun[i] = lrun[i] * corr + rsum;
      mrun[i] = nm;
      o0[i] *= corr; o1[i] *= corr; o2[i] *= corr; o3[i] *= corr;

      // stash P (bf16) in C-layout-addressed LDS for re-fragmenting as A.
      Pld[(half * 8 + i) * 32 + col]      = (bf16_t)p0;
      Pld[(half * 8 + i) * 32 + 16 + col] = (bf16_t)p1;
    }

    asm volatile("s_wait_dscnt 0x0" ::: "memory");  // LDS stores visible to loads

    v16bf pa = load_a_frag(Pld, 32);  // P as 16x32 A fragment
    v16bf bv0 = load_bt_frag(Vb + (size_t)0  * T_ + s0, T_);
    v16bf bv1 = load_bt_frag(Vb + (size_t)16 * T_ + s0, T_);
    v16bf bv2 = load_bt_frag(Vb + (size_t)32 * T_ + s0, T_);
    v16bf bv3 = load_bt_frag(Vb + (size_t)48 * T_ + s0, T_);
    o0 = wmma_bf16(pa, bv0, o0);
    o1 = wmma_bf16(pa, bv1, o1);
    o2 = wmma_bf16(pa, bv2, o2);
    o3 = wmma_bf16(pa, bv3, o3);
    asm volatile("s_wait_dscnt 0x0" ::: "memory");  // loads done before overwrite
  }

#pragma unroll
  for (int i = 0; i < 8; ++i) {
    float inv = 1.f / lrun[i];
    int rowg = q0 + half * 8 + i;
    size_t base = ((size_t)bb * T_ + rowg) * E_ + hh * D_;
    concat[base + 0  + col] = (bf16_t)(o0[i] * inv);
    concat[base + 16 + col] = (bf16_t)(o1[i] * inv);
    concat[base + 32 + col] = (bf16_t)(o2[i] * inv);
    concat[base + 48 + col] = (bf16_t)(o3[i] * inv);
  }
}

// ---------------------------------------------------------------------------
// Host-side orchestration
// ---------------------------------------------------------------------------

extern "C" void kernel_launch(void* const* d_in, const int* in_sizes, int n_in,
                              void* d_out, int out_size, void* d_ws,
                              size_t ws_size, hipStream_t stream) {
  const float* data = (const float*)d_in[0];
  const float* Wq   = (const float*)d_in[1];
  const float* Wk   = (const float*)d_in[2];
  const float* Wv   = (const float*)d_in[3];
  const float* Wfc  = (const float*)d_in[4];
  const float* bfc  = (const float*)d_in[5];
  const float* W1   = (const float*)d_in[6];
  const float* b1   = (const float*)d_in[7];
  const float* W2   = (const float*)d_in[8];
  const float* b2   = (const float*)d_in[9];
  const float* g1   = (const float*)d_in[10];
  const float* be1  = (const float*)d_in[11];
  const float* g2   = (const float*)d_in[12];
  const float* be2  = (const float*)d_in[13];
  float* out = (float*)d_out;

  char* ws = (char*)d_ws;
  size_t off = 0;
  auto alloc = [&](size_t bytes) -> void* {
    void* p = ws + off;
    off += (bytes + 255) & ~(size_t)255;
    return p;
  };

  bf16_t* hb     = (bf16_t*)alloc((size_t)M_ * E_ * 2);      // LN1 out
  bf16_t* Wqkv_t = (bf16_t*)alloc((size_t)3 * E_ * E_ * 2);  // [3072][1024]
  bf16_t* Wfct   = (bf16_t*)alloc((size_t)E_ * E_ * 2);
  bf16_t* W1t    = (bf16_t*)alloc((size_t)F_ * E_ * 2);      // [4096][1024]
  bf16_t* W2t    = (bf16_t*)alloc((size_t)E_ * F_ * 2);      // [1024][4096]
  bf16_t* qb     = (bf16_t*)alloc((size_t)B_ * H_ * T_ * D_ * 2);
  bf16_t* kb     = (bf16_t*)alloc((size_t)B_ * H_ * T_ * D_ * 2);
  bf16_t* vtb    = (bf16_t*)alloc((size_t)B_ * H_ * D_ * T_ * 2);
  bf16_t* concat = (bf16_t*)alloc((size_t)M_ * E_ * 2);
  float*  xres   = (float*)alloc((size_t)M_ * E_ * 4);       // x = data + attn
  bf16_t* h2     = (bf16_t*)alloc((size_t)M_ * E_ * 2);      // LN2 out
  bf16_t* a1     = (bf16_t*)alloc((size_t)M_ * F_ * 2);      // relu(h2 W1 + b1)
  (void)ws_size; (void)in_sizes; (void)n_in; (void)out_size;

  // 1. Weight conversion (fp32 -> bf16, transposed [N][K]).
  qkv_weights_to_bf16<<<dim3(3 * E_), dim3(256), 0, stream>>>(Wq, Wk, Wv, Wqkv_t);
  transpose_to_bf16<<<dim3(E_), dim3(256), 0, stream>>>(Wfc, Wfct, E_, E_);
  transpose_to_bf16<<<dim3(F_), dim3(256), 0, stream>>>(W1, W1t, E_, F_);
  transpose_to_bf16<<<dim3(E_), dim3(256), 0, stream>>>(W2, W2t, F_, E_);

  // 2. LN1: data -> hb (bf16).
  layernorm_bf16<<<dim3(M_), dim3(256), 0, stream>>>(data, g1, be1, hb);

  // 3. Fused QKV projection: [4096,1024] x [1024,3072] -> q/k/vT (mode 1).
  gemm_wmma<<<dim3(3 * E_ / 64, M_ / 128), dim3(256), 0, stream>>>(
      hb, Wqkv_t, 3 * E_, E_, nullptr, nullptr, nullptr, nullptr,
      qb, kb, vtb, 1);

  // 4. Flash attention -> concat [B*T][E] bf16.
  flash_attn<<<dim3(B_ * H_ * (T_ / 16)), dim3(32), 0, stream>>>(
      qb, kb, vtb, concat);

  // 5. Out projection + bias + residual(data) -> xres fp32 (mode 2).
  gemm_wmma<<<dim3(E_ / 64, M_ / 128), dim3(256), 0, stream>>>(
      concat, Wfct, E_, E_, bfc, data, xres, nullptr,
      nullptr, nullptr, nullptr, 2);

  // 6. LN2: xres -> h2 (bf16).
  layernorm_bf16<<<dim3(M_), dim3(256), 0, stream>>>(xres, g2, be2, h2);

  // 7. FFN1: relu(h2 W1 + b1) -> a1 bf16 (mode 3).
  gemm_wmma<<<dim3(F_ / 64, M_ / 128), dim3(256), 0, stream>>>(
      h2, W1t, F_, E_, b1, nullptr, nullptr, a1,
      nullptr, nullptr, nullptr, 3);

  // 8. FFN2 + bias + residual(xres) -> d_out fp32 (mode 2).
  gemm_wmma<<<dim3(E_ / 64, M_ / 128), dim3(256), 0, stream>>>(
      a1, W2t, E_, F_, b2, xres, out, nullptr,
      nullptr, nullptr, nullptr, 2);
}
